// Model_83141976916116
// MI455X (gfx1250) — compile-verified, matrix operand order checked by reading, not run on hardware
//
#include <hip/hip_runtime.h>
#include <hip/hip_bf16.h>

// ---------------------------------------------------------------------------
// GRU (6 layers, T=512, B=256, IN=128, H=256) on gfx1250.
// bf16 V_WMMA_F32_16X16X32_BF16 with fp32 accumulation for both the big input
// GEMM (131072 x 768 x K) and the 512 sequential recurrent GEMMs per layer
// (256 x 768 x 256); fused gate math in the step kernel.
// Input GEMM uses a 32x32 per-wave tile (2x2 WMMA) to halve the load-per-WMMA
// ratio (32B of operands per lane per 8192-MAC WMMA instead of 64B).
// ---------------------------------------------------------------------------

#define SEQ  512
#define BATCH 256
#define INDIM 128
#define HDIM 256
#define NLAYER 6
#define G3H  768                      // 3*H
#define MROWS (SEQ * BATCH)           // 131072

typedef __attribute__((ext_vector_type(16))) __bf16        v16bf;
typedef __attribute__((ext_vector_type(8)))  float         v8f;
typedef __attribute__((ext_vector_type(4)))  unsigned int  v4u;

union Frag {                 // 16 bf16 values = one WMMA A/B operand (32B)
    v4u   q[2];
    v16bf v;
};

__device__ __forceinline__ unsigned short f32_to_bf16(float f) {
    unsigned int u = __float_as_uint(f);
    unsigned int r = (u + 0x7FFFu + ((u >> 16) & 1u)) >> 16;   // RNE
    return (unsigned short)r;
}

// --------------------------- fp32 -> bf16 copy ------------------------------
__global__ void f2bf_kernel(const float* __restrict__ in,
                            unsigned short* __restrict__ out, int n) {
    int i = blockIdx.x * blockDim.x + threadIdx.x;
    if (i < n) out[i] = f32_to_bf16(in[i]);
}

// --------------------------- input GEMM -------------------------------------
// C[M, 768] = A[M, K] * B[768, K]^T   (A = bf16 activations, B = bf16 w_ih)
// 1 wave -> 32x32 C tile via 2x2 WMMA (4 accumulators, A/B fragments reused).
// block = 8 waves (2x4) = 64(M) x 128(N) tile.
__global__ void gemm_in_kernel(const unsigned short* __restrict__ A,
                               const unsigned short* __restrict__ Bw,
                               float* __restrict__ C, int K) {
    const int N = G3H;
    int tid  = threadIdx.x;
    int wave = tid >> 5;
    int lane = tid & 31;
    int l16  = lane & 15;
    int lhi  = lane >> 4;

    int row0 = blockIdx.y * 64 + (wave >> 2) * 32;   // M tile base
    int col0 = blockIdx.x * 128 + (wave & 3) * 32;   // N tile base

    const unsigned short* aRow0 = A  + (size_t)(row0 + l16) * K;
    const unsigned short* aRow1 = A  + (size_t)(row0 + 16 + l16) * K;
    const unsigned short* bRow0 = Bw + (size_t)(col0 + l16) * K;
    const unsigned short* bRow1 = Bw + (size_t)(col0 + 16 + l16) * K;

    v8f acc00 = {}, acc01 = {}, acc10 = {}, acc11 = {};
    for (int k = 0; k < K; k += 32) {
        int kb = k + lhi * 8;                        // CDNA5 16-bit frag layout
        Frag a0, a1, b0, b1;
        a0.q[0] = *(const v4u*)(aRow0 + kb);
        a0.q[1] = *(const v4u*)(aRow0 + kb + 16);
        a1.q[0] = *(const v4u*)(aRow1 + kb);
        a1.q[1] = *(const v4u*)(aRow1 + kb + 16);
        b0.q[0] = *(const v4u*)(bRow0 + kb);
        b0.q[1] = *(const v4u*)(bRow0 + kb + 16);
        b1.q[0] = *(const v4u*)(bRow1 + kb);
        b1.q[1] = *(const v4u*)(bRow1 + kb + 16);
        acc00 = __builtin_amdgcn_wmma_f32_16x16x32_bf16(
            false, a0.v, false, b0.v, (short)0, acc00, false, false);
        acc01 = __builtin_amdgcn_wmma_f32_16x16x32_bf16(
            false, a0.v, false, b1.v, (short)0, acc01, false, false);
        acc10 = __builtin_amdgcn_wmma_f32_16x16x32_bf16(
            false, a1.v, false, b0.v, (short)0, acc10, false, false);
        acc11 = __builtin_amdgcn_wmma_f32_16x16x32_bf16(
            false, a1.v, false, b1.v, (short)0, acc11, false, false);
    }

    // C/D layout: VGPR j -> (m = tile_m + 8*lhi + j, n = tile_n + l16)
    float* c00 = C + (size_t)(row0 + lhi * 8) * N + col0 + l16;
    float* c01 = c00 + 16;
    float* c10 = C + (size_t)(row0 + 16 + lhi * 8) * N + col0 + l16;
    float* c11 = c10 + 16;
#pragma unroll
    for (int j = 0; j < 8; ++j) {
        c00[(size_t)j * N] = acc00[j];
        c01[(size_t)j * N] = acc01[j];
        c10[(size_t)j * N] = acc10[j];
        c11[(size_t)j * N] = acc11[j];
    }
}

// --------------------------- recurrent step ---------------------------------
// gh = h_prev * w_hh^T (three gate column blocks), then fused GRU gates.
// 1 wave -> one 16(b) x 16(h) h_new patch with 3 accumulators (A reused 3x).
// grid = (H/128, B/16), block = 256 threads (8 waves along h).
__global__ void gru_step_kernel(const float* __restrict__ gi_t,          // (B, 768)
                                const unsigned short* __restrict__ whh,  // (768, H) bf16
                                const float* __restrict__ b_ih,          // (768)
                                const float* __restrict__ b_hh,          // (768)
                                const float* __restrict__ hprev_f,       // (B, H)
                                const unsigned short* __restrict__ hprev_bf,
                                float* __restrict__ hnew_f,
                                unsigned short* __restrict__ hnew_bf,
                                unsigned short* __restrict__ seq_bf,     // (B, H) slice
                                float* __restrict__ out_f) {             // (B, H) slice or null
    int tid  = threadIdx.x;
    int wave = tid >> 5;
    int lane = tid & 31;
    int l16  = lane & 15;
    int lhi  = lane >> 4;

    int b0 = blockIdx.y * 16;
    int n0 = blockIdx.x * 128 + wave * 16;

    const unsigned short* aRow = hprev_bf + (size_t)(b0 + l16) * HDIM;
    const unsigned short* bR   = whh + (size_t)(0 * HDIM + n0 + l16) * HDIM;
    const unsigned short* bZ   = whh + (size_t)(1 * HDIM + n0 + l16) * HDIM;
    const unsigned short* bN   = whh + (size_t)(2 * HDIM + n0 + l16) * HDIM;

    v8f accR = {}, accZ = {}, accN = {};
    for (int k = 0; k < HDIM; k += 32) {
        int kb = k + lhi * 8;
        Frag a, fr, fz, fn;
        a.q[0]  = *(const v4u*)(aRow + kb);
        a.q[1]  = *(const v4u*)(aRow + kb + 16);
        fr.q[0] = *(const v4u*)(bR + kb);
        fr.q[1] = *(const v4u*)(bR + kb + 16);
        fz.q[0] = *(const v4u*)(bZ + kb);
        fz.q[1] = *(const v4u*)(bZ + kb + 16);
        fn.q[0] = *(const v4u*)(bN + kb);
        fn.q[1] = *(const v4u*)(bN + kb + 16);
        accR = __builtin_amdgcn_wmma_f32_16x16x32_bf16(
            false, a.v, false, fr.v, (short)0, accR, false, false);
        accZ = __builtin_amdgcn_wmma_f32_16x16x32_bf16(
            false, a.v, false, fz.v, (short)0, accZ, false, false);
        accN = __builtin_amdgcn_wmma_f32_16x16x32_bf16(
            false, a.v, false, fn.v, (short)0, accN, false, false);
    }

    int col = n0 + l16;
    float bhr = b_hh[col], bhz = b_hh[HDIM + col], bhn = b_hh[2 * HDIM + col];
    float bir = b_ih[col], biz = b_ih[HDIM + col], bin_ = b_ih[2 * HDIM + col];

#pragma unroll
    for (int j = 0; j < 8; ++j) {
        int b = b0 + lhi * 8 + j;
        const float* g = gi_t + (size_t)b * G3H;
        float ir  = g[col] + bir;
        float iz  = g[HDIM + col] + biz;
        float inn = g[2 * HDIM + col] + bin_;
        float hr  = accR[j] + bhr;
        float hz  = accZ[j] + bhz;
        float hn  = accN[j] + bhn;

        float r = 1.0f / (1.0f + __expf(-(ir + hr)));
        float z = 1.0f / (1.0f + __expf(-(iz + hz)));
        float n = tanhf(inn + r * hn);

        size_t idx = (size_t)b * HDIM + col;
        float hp = hprev_f[idx];
        float ho = (1.0f - z) * n + z * hp;

        hnew_f[idx] = ho;
        unsigned short hb = f32_to_bf16(ho);
        hnew_bf[idx] = hb;
        seq_bf[idx]  = hb;
        if (out_f) out_f[idx] = ho;
    }
}

// ---------------------------------------------------------------------------
extern "C" void kernel_launch(void* const* d_in, const int* in_sizes, int n_in,
                              void* d_out, int out_size, void* d_ws, size_t ws_size,
                              hipStream_t stream) {
    const float* x        = (const float*)d_in[0];  // (512,256,128)
    const float* h0       = (const float*)d_in[1];  // (6,256,256)
    const float* w_ih0    = (const float*)d_in[2];  // (768,128)
    const float* w_ih_r   = (const float*)d_in[3];  // (5,768,256)
    const float* w_hh     = (const float*)d_in[4];  // (6,768,256)
    const float* b_ih     = (const float*)d_in[5];  // (6,768)
    const float* b_hh     = (const float*)d_in[6];  // (6,768)
    float* out            = (float*)d_out;          // (512,256,256)

    // ---- workspace layout (bytes) ----
    char* ws = (char*)d_ws;
    size_t off = 0;
    unsigned short* wih0_bf = (unsigned short*)(ws + off); off += (size_t)G3H * INDIM * 2;         // 192 KB
    unsigned short* wihr_bf = (unsigned short*)(ws + off); off += (size_t)5 * G3H * HDIM * 2;      // 1.9 MB
    unsigned short* whh_bf  = (unsigned short*)(ws + off); off += (size_t)NLAYER * G3H * HDIM * 2; // 2.3 MB
    unsigned short* h0_bf   = (unsigned short*)(ws + off); off += (size_t)NLAYER * BATCH * HDIM * 2;
    float*          hf0     = (float*)(ws + off);          off += (size_t)BATCH * HDIM * 4;
    float*          hf1     = (float*)(ws + off);          off += (size_t)BATCH * HDIM * 4;
    unsigned short* hbf0    = (unsigned short*)(ws + off); off += (size_t)BATCH * HDIM * 2;
    unsigned short* hbf1    = (unsigned short*)(ws + off); off += (size_t)BATCH * HDIM * 2;
    unsigned short* act_bf  = (unsigned short*)(ws + off); off += (size_t)MROWS * HDIM * 2;        // 67 MB
    float*          gi      = (float*)(ws + off);          off += (size_t)MROWS * G3H * 4;         // 403 MB
    (void)ws_size; (void)in_sizes; (void)n_in; (void)out_size;

    // ---- one-time fp32 -> bf16 conversions ----
    {
        int n;
        n = MROWS * INDIM;               // x -> act_bf (layer 0 input, K=128 packing)
        f2bf_kernel<<<(n + 255) / 256, 256, 0, stream>>>(x, act_bf, n);
        n = G3H * INDIM;
        f2bf_kernel<<<(n + 255) / 256, 256, 0, stream>>>(w_ih0, wih0_bf, n);
        n = 5 * G3H * HDIM;
        f2bf_kernel<<<(n + 255) / 256, 256, 0, stream>>>(w_ih_r, wihr_bf, n);
        n = NLAYER * G3H * HDIM;
        f2bf_kernel<<<(n + 255) / 256, 256, 0, stream>>>(w_hh, whh_bf, n);
        n = NLAYER * BATCH * HDIM;
        f2bf_kernel<<<(n + 255) / 256, 256, 0, stream>>>(h0, h0_bf, n);
    }

    float*          hbuf_f[2]  = { hf0, hf1 };
    unsigned short* hbuf_bf[2] = { hbf0, hbf1 };

    for (int li = 0; li < NLAYER; ++li) {
        const int K = (li == 0) ? INDIM : HDIM;
        const unsigned short* wih = (li == 0) ? wih0_bf
                                              : wihr_bf + (size_t)(li - 1) * G3H * HDIM;

        // Big input GEMM over all timesteps: gi = act_bf * wih^T
        dim3 gIn(G3H / 128, MROWS / 64);
        gemm_in_kernel<<<gIn, 256, 0, stream>>>(act_bf, wih, gi, K);

        const unsigned short* whh_l = whh_bf + (size_t)li * G3H * HDIM;
        const float* bih_l = b_ih + (size_t)li * G3H;
        const float* bhh_l = b_hh + (size_t)li * G3H;
        const bool last = (li == NLAYER - 1);

        // Sequential scan: one launch per timestep (stream orders the recurrence).
        for (int t = 0; t < SEQ; ++t) {
            const float* hp_f = (t == 0) ? (h0 + (size_t)li * BATCH * HDIM)
                                         : hbuf_f[(t - 1) & 1];
            const unsigned short* hp_bf = (t == 0) ? (h0_bf + (size_t)li * BATCH * HDIM)
                                                   : hbuf_bf[(t - 1) & 1];
            dim3 gStep(HDIM / 128, BATCH / 16);
            gru_step_kernel<<<gStep, 256, 0, stream>>>(
                gi + (size_t)t * BATCH * G3H,
                whh_l, bih_l, bhh_l,
                hp_f, hp_bf,
                hbuf_f[t & 1], hbuf_bf[t & 1],
                act_bf + (size_t)t * BATCH * HDIM,            // next layer's input (bf16)
                last ? (out + (size_t)t * BATCH * HDIM) : (float*)nullptr);
        }
    }
}
